// SimpleMoE_81157702025363
// MI455X (gfx1250) — matrix operand order repaired and checked by reading, not compiled, not run on hardware
//
#include <hip/hip_runtime.h>
#include <hip/hip_bf16.h>

// ---------------------------------------------------------------------------
// SimpleMoE (top-1) for MI455X / gfx1250.
// Prep: x -> bf16; W1/W2 -> bf16 transposed [N][K] (so GEMM B-tiles are
//       contiguous row copies).
// Gate: logits/softmax/argmax/bucket (atomic append).
// FFN1: h = gelu(x@W1+b1) -> bf16, FFN2: out = (h@W2+b2)*score.
// GEMMs: 128x256 block tile, 8 wave32s, 64x64 wave tile (16 wmma/K-step),
//        double-buffered LDS fed by GLOBAL_LOAD_ASYNC_TO_LDS_B128 with
//        s_wait_asynccnt pipelining. v_wmma_f32_16x16x32_bf16 compute.
// ---------------------------------------------------------------------------

#define B_ 16384
#define D_ 1024
#define H_ 4096
#define E_ 8

#define BM 128          // block M tile
#define BNT 256         // block N tile
#define BK 32           // K step (one wmma)
#define LDSTRIDE 20     // dwords per LDS row (16 data + 4 pad, 16B aligned)

typedef __attribute__((ext_vector_type(16))) __bf16 v16bf;
typedef __attribute__((ext_vector_type(8)))  float  v8f;

union Frag16 { v16bf v; unsigned int u[8]; };

__device__ __forceinline__ unsigned short f32_to_bf16_rne(float f) {
    unsigned int u = __float_as_uint(f);
    u += 0x7FFFu + ((u >> 16) & 1u);
    return (unsigned short)(u >> 16);
}
__device__ __forceinline__ unsigned int pack_bf16x2(float lo, float hi) {
    return (unsigned int)f32_to_bf16_rne(lo) |
           ((unsigned int)f32_to_bf16_rne(hi) << 16);
}
__device__ __forceinline__ float gelu_tanh(float v) {
    float c = 0.7978845608028654f * (v + 0.044715f * v * v * v);
    return 0.5f * v * (1.0f + tanhf(c));
}

// Async global->LDS copy of 16 bytes (per-lane LDS dest + global addr).
__device__ __forceinline__ void async_b128(unsigned lds_byte, const void* g) {
    asm volatile("global_load_async_to_lds_b128 %0, %1, off"
                 :: "v"(lds_byte), "v"((unsigned long long)(uintptr_t)g)
                 : "memory");
}

// ---------------------------------------------------------------------------
// Prep kernels
// ---------------------------------------------------------------------------
__global__ __launch_bounds__(256) void cvt_x_kernel(
    const float* __restrict__ x, unsigned short* __restrict__ xb, size_t n4)
{
    size_t i = (size_t)blockIdx.x * 256 + threadIdx.x;
    size_t stride = (size_t)gridDim.x * 256;
    for (; i < n4; i += stride) {
        float4 a = ((const float4*)x)[i];
        uint2 r;
        r.x = pack_bf16x2(a.x, a.y);
        r.y = pack_bf16x2(a.z, a.w);
        ((uint2*)xb)[i] = r;
    }
}

// src [E][K][N] fp32 -> dst [E][N][K] bf16 (transpose + convert), 64x64 tiles.
__global__ __launch_bounds__(256) void transpose_cvt_kernel(
    const float* __restrict__ src, unsigned short* __restrict__ dst,
    int K, int N)
{
    __shared__ unsigned short tile[64][66];
    int e = blockIdx.z;
    const float* s = src + (size_t)e * K * N;
    unsigned short* d = dst + (size_t)e * K * N;
    int n0 = blockIdx.x * 64, k0 = blockIdx.y * 64;
    int t = threadIdx.x;
#pragma unroll
    for (int i = 0; i < 16; ++i) {
        int idx = t + i * 256;
        int kk = idx >> 6, nn = idx & 63;
        tile[kk][nn] = f32_to_bf16_rne(s[(size_t)(k0 + kk) * N + (n0 + nn)]);
    }
    __syncthreads();
#pragma unroll
    for (int i = 0; i < 16; ++i) {
        int idx = t + i * 256;
        int nn = idx >> 6, kk = idx & 63;
        d[(size_t)(n0 + nn) * K + (k0 + kk)] = tile[kk][nn];
    }
}

// ---------------------------------------------------------------------------
// Gating: one wave per token.
// ---------------------------------------------------------------------------
__global__ __launch_bounds__(256) void moe_gate_kernel(
    const float* __restrict__ x, const float* __restrict__ Wg,
    float* __restrict__ gate_scores, float* __restrict__ top_expert_out,
    float* __restrict__ top_score, int* __restrict__ counts,
    int* __restrict__ bucket)
{
    int wave = threadIdx.x >> 5;
    int lane = threadIdx.x & 31;
    int t = blockIdx.x * 8 + wave;
    if (t >= B_) return;

    const float* xr = x + (size_t)t * D_;
    float acc[E_];
#pragma unroll
    for (int e = 0; e < E_; ++e) acc[e] = 0.0f;

    for (int d = lane; d < D_; d += 32) {
        float xv = xr[d];
        const float4* wr = (const float4*)(Wg + (size_t)d * E_);
        float4 w0 = wr[0];
        float4 w1 = wr[1];
        acc[0] += xv * w0.x; acc[1] += xv * w0.y;
        acc[2] += xv * w0.z; acc[3] += xv * w0.w;
        acc[4] += xv * w1.x; acc[5] += xv * w1.y;
        acc[6] += xv * w1.z; acc[7] += xv * w1.w;
    }
#pragma unroll
    for (int e = 0; e < E_; ++e) {
#pragma unroll
        for (int off = 16; off > 0; off >>= 1)
            acc[e] += __shfl_down(acc[e], off, 32);
    }
    if (lane == 0) {
        float mx = acc[0];
        int best = 0;
#pragma unroll
        for (int e = 1; e < E_; ++e)
            if (acc[e] > mx) { mx = acc[e]; best = e; }
        float sum = 0.0f;
        float p[E_];
#pragma unroll
        for (int e = 0; e < E_; ++e) { p[e] = __expf(acc[e] - mx); sum += p[e]; }
        float inv = 1.0f / sum;
#pragma unroll
        for (int e = 0; e < E_; ++e)
            gate_scores[(size_t)t * E_ + e] = p[e] * inv;
        top_expert_out[t] = (float)best;
        top_score[t] = p[best] * inv;
        int slot = atomicAdd(&counts[best], 1);
        bucket[(size_t)best * B_ + slot] = t;
    }
}

// ---------------------------------------------------------------------------
// WMMA GEMM core (shared by both FFN stages via template-ish macro-free dup).
// A: bf16 gathered token rows [row][K]; B: bf16 transposed weights [N][K].
// ---------------------------------------------------------------------------
__device__ __forceinline__ void gemm_core(
    const unsigned short* __restrict__ Arows0,  // row base ptr for grow0 token
    const unsigned short* __restrict__ Arows1,  // row base ptr for grow1 token
    const unsigned short* __restrict__ Bbase,   // Wt + n-row base (tid-dependent)
    unsigned int (*As)[BM * LDSTRIDE],
    unsigned int (*Bs)[BNT * LDSTRIDE],
    int K, int tid, v8f acc[4][4])
{
    const int aseg = tid & 3;
    const unsigned lA0 = (unsigned)(uintptr_t)&As[0][((tid >> 2)) * LDSTRIDE + aseg * 4];
    const unsigned lA1 = (unsigned)(uintptr_t)&As[0][((tid >> 2) + 64) * LDSTRIDE + aseg * 4];
    const unsigned lB  = (unsigned)(uintptr_t)&Bs[0][((tid >> 2)) * LDSTRIDE + aseg * 4];
    const unsigned bufAbytes = (unsigned)(BM * LDSTRIDE * 4);
    const unsigned bufBbytes = (unsigned)(BNT * LDSTRIDE * 4);
    const unsigned rowBbytes = (unsigned)(64 * LDSTRIDE * 4);

    const int lane = tid & 31, wave = tid >> 5;
    const int waveM = wave >> 2, waveN = wave & 3;
    const int hi = lane >> 4, ln = lane & 15;

    const int KT = K / BK;

    // prefetch tile 0
    {
        async_b128(lA0, Arows0);
        async_b128(lA1, Arows1);
        async_b128(lB,                 Bbase);
        async_b128(lB + rowBbytes,     Bbase + (size_t)64 * K);
        async_b128(lB + 2 * rowBbytes, Bbase + (size_t)128 * K);
        async_b128(lB + 3 * rowBbytes, Bbase + (size_t)192 * K);
    }

    for (int kt = 0; kt < KT; ++kt) {
        int buf = kt & 1;
        if (kt + 1 < KT) {
            int kk = (kt + 1) * BK;
            unsigned ao = (buf ^ 1) ? bufAbytes : 0u;
            unsigned bo = (buf ^ 1) ? bufBbytes : 0u;
            async_b128(lA0 + ao, Arows0 + kk);
            async_b128(lA1 + ao, Arows1 + kk);
            async_b128(lB + bo,                 Bbase + kk);
            async_b128(lB + bo + rowBbytes,     Bbase + (size_t)64 * K + kk);
            async_b128(lB + bo + 2 * rowBbytes, Bbase + (size_t)128 * K + kk);
            async_b128(lB + bo + 3 * rowBbytes, Bbase + (size_t)192 * K + kk);
            asm volatile("s_wait_asynccnt 0x6" ::: "memory");
        } else {
            asm volatile("s_wait_asynccnt 0x0" ::: "memory");
        }
        __syncthreads();

        const unsigned int* Ab = As[buf];
        const unsigned int* Bb = Bs[buf];

        Frag16 afr[4];
#pragma unroll
        for (int fm = 0; fm < 4; ++fm) {
            const unsigned int* ar = &Ab[(waveM * 64 + fm * 16 + ln) * LDSTRIDE];
#pragma unroll
            for (int j = 0; j < 4; ++j) {
                afr[fm].u[j]     = ar[hi * 4 + j];
                afr[fm].u[4 + j] = ar[8 + hi * 4 + j];
            }
        }
#pragma unroll
        for (int fn = 0; fn < 4; ++fn) {
            Frag16 bfr;
            const unsigned int* br = &Bb[(waveN * 64 + fn * 16 + ln) * LDSTRIDE];
#pragma unroll
            for (int i = 0; i < 8; ++i) bfr.u[i] = br[(i >> 2) * 8 + hi * 4 + (i & 3)];
#pragma unroll
            for (int fm = 0; fm < 4; ++fm)
                acc[fm][fn] = __builtin_amdgcn_wmma_f32_16x16x32_bf16(
                    false, afr[fm].v, false, bfr.v,
                    (short)0, acc[fm][fn], false, false);
        }
        __syncthreads();
    }
}

// Stage 1: h = gelu( x_gather @ W1 + b1 ) -> hbf (bf16)
__global__ __launch_bounds__(256) void moe_ffn1_kernel(
    const unsigned short* __restrict__ xb, const unsigned short* __restrict__ W1t,
    const float* __restrict__ b1, const int* __restrict__ counts,
    const int* __restrict__ bucket, unsigned short* __restrict__ hbf)
{
    constexpr int K = D_;
    int e = blockIdx.z;
    int cnt = counts[e];
    int m0 = blockIdx.x * BM;
    if (m0 >= cnt) return;
    int n0 = blockIdx.y * BNT;

    const unsigned short* Wt = W1t + (size_t)e * K * H_;  // [H][D]
    const int* bkt = bucket + (size_t)e * B_;

    __shared__ alignas(16) unsigned int As[2][BM * LDSTRIDE];
    __shared__ alignas(16) unsigned int Bs[2][BNT * LDSTRIDE];

    int tid = threadIdx.x;
    int grow0 = m0 + (tid >> 2);
    int grow1 = grow0 + 64;
    int tok0 = bkt[min(grow0, cnt - 1)];
    int tok1 = bkt[min(grow1, cnt - 1)];
    const unsigned short* Arows0 = xb + (size_t)tok0 * K + (tid & 3) * 8;
    const unsigned short* Arows1 = xb + (size_t)tok1 * K + (tid & 3) * 8;
    const unsigned short* Bbase  = Wt + (size_t)(n0 + (tid >> 2)) * K + (tid & 3) * 8;

    v8f acc[4][4];
#pragma unroll
    for (int fm = 0; fm < 4; ++fm)
#pragma unroll
        for (int fn = 0; fn < 4; ++fn)
            acc[fm][fn] = (v8f){0.f, 0.f, 0.f, 0.f, 0.f, 0.f, 0.f, 0.f};

    gemm_core(Arows0, Arows1, Bbase, As, Bs, K, tid, acc);

    int lane = tid & 31, wave = tid >> 5;
    int waveM = wave >> 2, waveN = wave & 3;
    int hi = lane >> 4, ln = lane & 15;
#pragma unroll
    for (int fm = 0; fm < 4; ++fm) {
#pragma unroll
        for (int fn = 0; fn < 4; ++fn) {
            int n = n0 + waveN * 64 + fn * 16 + ln;
            float bias = b1[(size_t)e * H_ + n];
#pragma unroll
            for (int r = 0; r < 8; ++r) {
                int gm = m0 + waveM * 64 + fm * 16 + hi * 8 + r;
                if (gm < cnt) {
                    int tok = bkt[gm];
                    float v = acc[fm][fn][r] + bias;
                    hbf[(size_t)tok * H_ + n] = f32_to_bf16_rne(gelu_tanh(v));
                }
            }
        }
    }
}

// Stage 2: out = (h_gather @ W2 + b2) * top_score -> fp32 scatter.
__global__ __launch_bounds__(256) void moe_ffn2_kernel(
    const unsigned short* __restrict__ hbf, const unsigned short* __restrict__ W2t,
    const float* __restrict__ b2, const int* __restrict__ counts,
    const int* __restrict__ bucket, const float* __restrict__ top_score,
    float* __restrict__ out)
{
    constexpr int K = H_;
    int e = blockIdx.z;
    int cnt = counts[e];
    int m0 = blockIdx.x * BM;
    if (m0 >= cnt) return;
    int n0 = blockIdx.y * BNT;

    const unsigned short* Wt = W2t + (size_t)e * K * D_;  // [D][H]
    const int* bkt = bucket + (size_t)e * B_;

    __shared__ alignas(16) unsigned int As[2][BM * LDSTRIDE];
    __shared__ alignas(16) unsigned int Bs[2][BNT * LDSTRIDE];

    int tid = threadIdx.x;
    int grow0 = m0 + (tid >> 2);
    int grow1 = grow0 + 64;
    int tok0 = bkt[min(grow0, cnt - 1)];
    int tok1 = bkt[min(grow1, cnt - 1)];
    const unsigned short* Arows0 = hbf + (size_t)tok0 * K + (tid & 3) * 8;
    const unsigned short* Arows1 = hbf + (size_t)tok1 * K + (tid & 3) * 8;
    const unsigned short* Bbase  = Wt + (size_t)(n0 + (tid >> 2)) * K + (tid & 3) * 8;

    v8f acc[4][4];
#pragma unroll
    for (int fm = 0; fm < 4; ++fm)
#pragma unroll
        for (int fn = 0; fn < 4; ++fn)
            acc[fm][fn] = (v8f){0.f, 0.f, 0.f, 0.f, 0.f, 0.f, 0.f, 0.f};

    gemm_core(Arows0, Arows1, Bbase, As, Bs, K, tid, acc);

    int lane = tid & 31, wave = tid >> 5;
    int waveM = wave >> 2, waveN = wave & 3;
    int hi = lane >> 4, ln = lane & 15;
#pragma unroll
    for (int fm = 0; fm < 4; ++fm) {
#pragma unroll
        for (int fn = 0; fn < 4; ++fn) {
            int n = n0 + waveN * 64 + fn * 16 + ln;
            float bias = b2[(size_t)e * D_ + n];
#pragma unroll
            for (int r = 0; r < 8; ++r) {
                int gm = m0 + waveM * 64 + fm * 16 + hi * 8 + r;
                if (gm < cnt) {
                    int tok = bkt[gm];
                    float v = (acc[fm][fn][r] + bias) * top_score[tok];
                    out[(size_t)tok * D_ + n] = v;
                }
            }
        }
    }
}

__global__ void moe_density_kernel(const int* __restrict__ counts,
                                   float* __restrict__ density)
{
    int e = threadIdx.x;
    if (e < E_) density[e] = (float)counts[e] * (1.0f / (float)B_);
}

// ---------------------------------------------------------------------------
// Workspace layout (bytes)
// ---------------------------------------------------------------------------
static const size_t SZ_XB  = (size_t)B_ * D_ * 2;            // 32 MiB
static const size_t SZ_WB  = (size_t)E_ * D_ * H_ * 2;       // 64 MiB each
static const size_t SZ_HBF = (size_t)B_ * H_ * 2;            // 128 MiB
static const size_t OFF_XB  = 0;
static const size_t OFF_W1T = OFF_XB + SZ_XB;
static const size_t OFF_W2T = OFF_W1T + SZ_WB;
static const size_t OFF_HBF = OFF_W2T + SZ_WB;
static const size_t OFF_BKT = OFF_HBF + SZ_HBF;
static const size_t SZ_BKT  = (size_t)E_ * B_ * 4;
static const size_t OFF_CNT = OFF_BKT + SZ_BKT;
static const size_t OFF_TS  = OFF_CNT + 256;

extern "C" void kernel_launch(void* const* d_in, const int* in_sizes, int n_in,
                              void* d_out, int out_size, void* d_ws, size_t ws_size,
                              hipStream_t stream) {
    const float* x  = (const float*)d_in[0];
    const float* Wg = (const float*)d_in[1];
    const float* W1 = (const float*)d_in[2];
    const float* b1 = (const float*)d_in[3];
    const float* W2 = (const float*)d_in[4];
    const float* b2 = (const float*)d_in[5];

    char* ws = (char*)d_ws;
    unsigned short* xb  = (unsigned short*)(ws + OFF_XB);
    unsigned short* w1t = (unsigned short*)(ws + OFF_W1T);
    unsigned short* w2t = (unsigned short*)(ws + OFF_W2T);
    unsigned short* hbf = (unsigned short*)(ws + OFF_HBF);
    int*   bucket = (int*)(ws + OFF_BKT);
    int*   counts = (int*)(ws + OFF_CNT);
    float* tscore = (float*)(ws + OFF_TS);

    float* out  = (float*)d_out;                       // [B, D]
    float* gs   = out + (size_t)B_ * D_;               // [B, E]
    float* te   = gs + (size_t)B_ * E_;                // [B]
    float* dens = te + B_;                             // [E]

    hipMemsetAsync(counts, 0, 256, stream);

    cvt_x_kernel<<<4096, 256, 0, stream>>>(x, xb, (size_t)B_ * D_ / 4);

    dim3 gt1(H_ / 64, D_ / 64, E_);
    transpose_cvt_kernel<<<gt1, 256, 0, stream>>>(W1, w1t, D_, H_);
    dim3 gt2(D_ / 64, H_ / 64, E_);
    transpose_cvt_kernel<<<gt2, 256, 0, stream>>>(W2, w2t, H_, D_);

    moe_gate_kernel<<<B_ / 8, 256, 0, stream>>>(x, Wg, gs, te, tscore, counts,
                                                bucket);

    dim3 g1(B_ / BM, H_ / BNT, E_);
    moe_ffn1_kernel<<<g1, 256, 0, stream>>>(xb, w1t, b1, counts, bucket, hbf);

    dim3 g2(B_ / BM, D_ / BNT, E_);
    moe_ffn2_kernel<<<g2, 256, 0, stream>>>(hbf, w2t, b2, counts, bucket,
                                            tscore, out);

    moe_density_kernel<<<1, 32, 0, stream>>>(counts, dens);
}
